// SubstrateGCN_74217034875250
// MI455X (gfx1250) — compile-verified, matrix operand order checked by reading, not run on hardware
//
#include <hip/hip_runtime.h>

#define N_NODESC 50000
#define N_EDGESC 800000
#define N_ETOT   850000   // edges + self loops
#define NUM_G    8
#define HEADS    4
#define CHN      64
#define HC       256      // HEADS*CHN
#define EPSV     1e-5f
#define SLOPE    0.2f

typedef __attribute__((ext_vector_type(16))) _Float16 v16h;
typedef __attribute__((ext_vector_type(8)))  _Float16 v8h;
typedef __attribute__((ext_vector_type(8)))  float    v8f;

__device__ __forceinline__ unsigned f2o(float f) {
  unsigned u = __float_as_uint(f);
  return (u & 0x80000000u) ? ~u : (u | 0x80000000u);
}
__device__ __forceinline__ float o2f(unsigned u) {
  return (u & 0x80000000u) ? __uint_as_float(u ^ 0x80000000u) : __uint_as_float(~u);
}
__device__ __forceinline__ float waveAllSum(float v) {
#pragma unroll
  for (int m = 1; m < 32; m <<= 1) v += __shfl_xor(v, m, 32);
  return v;
}

// ---------------- self-loop edge-attr (mean of in-edge attrs) ----------------
__global__ void selfloop_accum(const long long* __restrict__ ei,
                               const float* __restrict__ ea,
                               float* __restrict__ lsum, float* __restrict__ cntf) {
  int e = blockIdx.x * blockDim.x + threadIdx.x;
  if (e >= N_EDGESC) return;
  int t = (int)ei[N_EDGESC + e];
  atomicAdd(&lsum[t * 2 + 0], ea[e * 2 + 0]);
  atomicAdd(&lsum[t * 2 + 1], ea[e * 2 + 1]);
  atomicAdd(&cntf[t], 1.0f);
}

__global__ void selfloop_finalize(const float* __restrict__ lsum,
                                  const float* __restrict__ cntf,
                                  float* __restrict__ lattr) {
  int i = blockIdx.x * blockDim.x + threadIdx.x;
  if (i >= N_NODESC) return;
  float inv = 1.0f / fmaxf(cntf[i], 1.0f);
  lattr[i * 2 + 0] = lsum[i * 2 + 0] * inv;
  lattr[i * 2 + 1] = lsum[i * 2 + 1] * inv;
}

// ---------------- conv1 node transforms (K=5, VALU) ----------------
__global__ void conv1_transform(const float* __restrict__ x,
                                const float* __restrict__ Wl, const float* __restrict__ bl,
                                const float* __restrict__ Wr, const float* __restrict__ br,
                                float* __restrict__ Xl, float* __restrict__ Xr) {
  int idx = blockIdx.x * blockDim.x + threadIdx.x;
  if (idx >= N_NODESC * HC) return;
  int i = idx >> 8, c = idx & 255;
  const float* xi = x + (size_t)i * 5;
  float sl = bl[c], sr = br[c];
#pragma unroll
  for (int k = 0; k < 5; ++k) {
    float v = xi[k];
    sl += v * Wl[k * HC + c];
    sr += v * Wr[k * HC + c];
  }
  Xl[idx] = sl;
  Xr[idx] = sr;
}

// ---------------- weight prep: transpose + f32->f16 (one-time, 256x256 x2) ----------------
__global__ void convert_w_f16(const float* __restrict__ Wl, const float* __restrict__ Wr,
                              _Float16* __restrict__ Wlt, _Float16* __restrict__ Wrt) {
  int idx = blockIdx.x * blockDim.x + threadIdx.x;   // [n][k] output layout
  if (idx >= HC * HC) return;
  int n = idx >> 8, k = idx & 255;
  Wlt[idx] = (_Float16)Wl[k * HC + n];
  Wrt[idx] = (_Float16)Wr[k * HC + n];
}

// ---------------- conv2 transforms: dual 256x256 GEMM, WMMA f16, 4 N-tiles/wave ----------------
__global__ void gemm_dual_256(const _Float16* __restrict__ Hf,
                              const _Float16* __restrict__ Wlt, const float* __restrict__ bl,
                              const _Float16* __restrict__ Wrt, const float* __restrict__ br,
                              float* __restrict__ Xl, float* __restrict__ Xr) {
  int gid  = blockIdx.x * blockDim.x + threadIdx.x;
  int wave = gid >> 5;
  int lane = gid & 31;
  int tileM = wave >> 2;        // 3125 row tiles (50000 = 3125*16 exactly)
  int ng    = wave & 3;         // group of 4 column tiles
  if (tileM >= N_NODESC / 16) return;

  int rowA  = tileM * 16 + (lane & 15);
  int laneN = lane & 15;
  int khalf = (lane < 16) ? 0 : 8;

  v8f accl[4] = {};
  v8f accr[4] = {};
  for (int k0 = 0; k0 < HC; k0 += 32) {
    const v8h* pa = (const v8h*)(Hf + (size_t)rowA * HC + k0 + khalf);
    v8h a0 = pa[0], a1 = pa[2];     // K = khalf..+8 and khalf+16..+24
    v16h a = __builtin_shufflevector(a0, a1, 0, 1, 2, 3, 4, 5, 6, 7,
                                             8, 9, 10, 11, 12, 13, 14, 15);
#pragma unroll
    for (int t = 0; t < 4; ++t) {
      int col = (ng * 4 + t) * 16 + laneN;
      const v8h* pl = (const v8h*)(Wlt + (size_t)col * HC + k0 + khalf);
      const v8h* pr = (const v8h*)(Wrt + (size_t)col * HC + k0 + khalf);
      v8h l0 = pl[0], l1 = pl[2];
      v8h r0 = pr[0], r1 = pr[2];
      v16h bL = __builtin_shufflevector(l0, l1, 0, 1, 2, 3, 4, 5, 6, 7,
                                                8, 9, 10, 11, 12, 13, 14, 15);
      v16h bR = __builtin_shufflevector(r0, r1, 0, 1, 2, 3, 4, 5, 6, 7,
                                                8, 9, 10, 11, 12, 13, 14, 15);
      accl[t] = __builtin_amdgcn_wmma_f32_16x16x32_f16(false, a, false, bL, (short)0, accl[t], false, false);
      accr[t] = __builtin_amdgcn_wmma_f32_16x16x32_f16(false, a, false, bR, (short)0, accr[t], false, false);
    }
  }
  int rbase = tileM * 16 + ((lane < 16) ? 0 : 8);
#pragma unroll
  for (int t = 0; t < 4; ++t) {
    int col = (ng * 4 + t) * 16 + laneN;
    float ba = bl[col], bb = br[col];
#pragma unroll
    for (int r = 0; r < 8; ++r) {
      size_t o = (size_t)(rbase + r) * HC + col;
      Xl[o] = accl[t][r] + ba;
      Xr[o] = accr[t][r] + bb;
    }
  }
}

// ---------------- edge logits + segment max (wave per edge) ----------------
__global__ void edge_logits(const float* __restrict__ Xl, const float* __restrict__ Xr,
                            const float* __restrict__ We,   // [2,256]
                            const float* __restrict__ att,  // [4,64] flat
                            const long long* __restrict__ ei,
                            const float* __restrict__ ea,
                            const float* __restrict__ lattr,
                            float* __restrict__ logits, unsigned* __restrict__ lmax) {
  int gid = blockIdx.x * blockDim.x + threadIdx.x;
  int e = gid >> 5, lane = gid & 31;
  if (e >= N_ETOT) return;
  int src, tgt; float ea0, ea1;
  if (e < N_EDGESC) {
    src = (int)ei[e];
    tgt = (int)ei[N_EDGESC + e];
    ea0 = ea[e * 2 + 0]; ea1 = ea[e * 2 + 1];
  } else {
    src = tgt = e - N_EDGESC;
    ea0 = lattr[src * 2 + 0]; ea1 = lattr[src * 2 + 1];
  }
  int ch0 = lane * 8;
  const float4* pl = (const float4*)(Xl + (size_t)src * HC + ch0);
  const float4* pr = (const float4*)(Xr + (size_t)tgt * HC + ch0);
  float4 l0 = pl[0], l1 = pl[1], r0 = pr[0], r1 = pr[1];
  float lv[8] = {l0.x, l0.y, l0.z, l0.w, l1.x, l1.y, l1.z, l1.w};
  float rv[8] = {r0.x, r0.y, r0.z, r0.w, r1.x, r1.y, r1.z, r1.w};
  float part = 0.0f;
#pragma unroll
  for (int j = 0; j < 8; ++j) {
    int ch = ch0 + j;
    float m = lv[j] + rv[j] + ea0 * We[ch] + ea1 * We[HC + ch];
    m = (m > 0.0f) ? m : SLOPE * m;
    part += att[ch] * m;
  }
  // reduce within 8-lane groups (one head per group)
  part += __shfl_xor(part, 1, 32);
  part += __shfl_xor(part, 2, 32);
  part += __shfl_xor(part, 4, 32);
  if ((lane & 7) == 0) {
    int h = lane >> 3;
    logits[(size_t)e * HEADS + h] = part;
    atomicMax(&lmax[tgt * HEADS + h], f2o(part));
  }
}

// ---------------- exp(logit - max) + segment sum ----------------
__global__ void edge_exp(const long long* __restrict__ ei,
                         const unsigned* __restrict__ lmax,
                         float* __restrict__ logits, float* __restrict__ denom) {
  int t = blockIdx.x * blockDim.x + threadIdx.x;
  if (t >= N_ETOT * HEADS) return;
  int e = t >> 2, h = t & 3;
  int tgt = (e < N_EDGESC) ? (int)ei[N_EDGESC + e] : (e - N_EDGESC);
  float a = __expf(logits[t] - o2f(lmax[tgt * HEADS + h]));
  logits[t] = a;
  atomicAdd(&denom[tgt * HEADS + h], a);
}

// ---------------- weighted scatter-add (wave per edge) ----------------
__global__ void edge_scatter(const float* __restrict__ Xl,
                             const long long* __restrict__ ei,
                             const float* __restrict__ abuf,
                             const float* __restrict__ denom,
                             float* __restrict__ agg) {
  int gid = blockIdx.x * blockDim.x + threadIdx.x;
  int e = gid >> 5, lane = gid & 31;
  if (e >= N_ETOT) return;
  int src, tgt;
  if (e < N_EDGESC) { src = (int)ei[e]; tgt = (int)ei[N_EDGESC + e]; }
  else              { src = tgt = e - N_EDGESC; }
  int h = lane >> 3;   // lane's 8 channels all belong to one head
  float alpha = abuf[(size_t)e * HEADS + h] / denom[tgt * HEADS + h];
  int ch0 = lane * 8;
  const float4* pl = (const float4*)(Xl + (size_t)src * HC + ch0);
  float4 l0 = pl[0], l1 = pl[1];
  float lv[8] = {l0.x, l0.y, l0.z, l0.w, l1.x, l1.y, l1.z, l1.w};
  float* dst = agg + (size_t)tgt * HC + ch0;
#pragma unroll
  for (int j = 0; j < 8; ++j) atomicAdd(&dst[j], lv[j] * alpha);
}

// ---------------- post conv1: +bias, LayerNorm(256), ReLU -> f16 for WMMA ----------------
__global__ void node_post1(const float* __restrict__ agg, const float* __restrict__ bias,
                           const float* __restrict__ lnw, const float* __restrict__ lnb,
                           _Float16* __restrict__ hout) {
  int gid = blockIdx.x * blockDim.x + threadIdx.x;
  int i = gid >> 5, lane = gid & 31;
  if (i >= N_NODESC) return;
  int ch0 = lane * 8;
  float v[8]; float s = 0.0f;
#pragma unroll
  for (int j = 0; j < 8; ++j) { v[j] = agg[(size_t)i * HC + ch0 + j] + bias[ch0 + j]; s += v[j]; }
  float mu = waveAllSum(s) * (1.0f / HC);
  float q = 0.0f;
#pragma unroll
  for (int j = 0; j < 8; ++j) { float d = v[j] - mu; q += d * d; }
  float var = waveAllSum(q) * (1.0f / HC);
  float rs = rsqrtf(var + EPSV);
#pragma unroll
  for (int j = 0; j < 8; ++j) {
    int ch = ch0 + j;
    float y = (v[j] - mu) * rs * lnw[ch] + lnb[ch];
    hout[(size_t)i * HC + ch] = (_Float16)fmaxf(y, 0.0f);
  }
}

// ---------------- post conv2: head-mean, +bias, LayerNorm(64), ReLU ----------------
__global__ void node_post2(const float* __restrict__ agg, const float* __restrict__ bias,
                           const float* __restrict__ lnw, const float* __restrict__ lnb,
                           float* __restrict__ hout) {
  int gid = blockIdx.x * blockDim.x + threadIdx.x;
  int i = gid >> 5, lane = gid & 31;
  if (i >= N_NODESC) return;
  float v[2]; float s = 0.0f;
#pragma unroll
  for (int j = 0; j < 2; ++j) {
    int c = lane * 2 + j;
    const float* a = agg + (size_t)i * HC;
    v[j] = 0.25f * (a[c] + a[CHN + c] + a[2 * CHN + c] + a[3 * CHN + c]) + bias[c];
    s += v[j];
  }
  float mu = waveAllSum(s) * (1.0f / CHN);
  float q = 0.0f;
#pragma unroll
  for (int j = 0; j < 2; ++j) { float d = v[j] - mu; q += d * d; }
  float var = waveAllSum(q) * (1.0f / CHN);
  float rs = rsqrtf(var + EPSV);
#pragma unroll
  for (int j = 0; j < 2; ++j) {
    int c = lane * 2 + j;
    float y = (v[j] - mu) * rs * lnw[c] + lnb[c];
    hout[(size_t)i * CHN + c] = fmaxf(y, 0.0f);
  }
}

// ---------------- graph pooling ----------------
__global__ void pool_nodes(const float* __restrict__ h2, const long long* __restrict__ batch,
                           float* __restrict__ gsum, unsigned* __restrict__ gmax,
                           float* __restrict__ gcnt) {
  int gid = blockIdx.x * blockDim.x + threadIdx.x;
  int i = gid >> 5, lane = gid & 31;
  if (i >= N_NODESC) return;
  int b = (int)batch[i];
#pragma unroll
  for (int j = 0; j < 2; ++j) {
    int c = lane * 2 + j;
    float v = h2[(size_t)i * CHN + c];
    atomicAdd(&gsum[b * CHN + c], v);
    atomicMax(&gmax[b * CHN + c], f2o(v));
  }
  if (lane == 0) atomicAdd(&gcnt[b], 1.0f);
}

// ---------------- final projection: (8x128) @ (128x128) + b ----------------
__global__ void final_proj(const float* __restrict__ gsum, const unsigned* __restrict__ gmax,
                           const float* __restrict__ gcnt,
                           const float* __restrict__ projW, const float* __restrict__ projb,
                           float* __restrict__ out) {
  __shared__ float g[NUM_G * 128];
  int t = threadIdx.x;                 // 1024 threads
  if (t < NUM_G * CHN) {
    int b = t >> 6, c = t & 63;
    g[b * 128 + c]       = gsum[b * CHN + c] / fmaxf(gcnt[b], 1.0f);
    g[b * 128 + 64 + c]  = o2f(gmax[b * CHN + c]);
  }
  __syncthreads();
  int b = t >> 7, c = t & 127;
  float s = projb[c];
#pragma unroll 4
  for (int k = 0; k < 128; ++k) s += g[b * 128 + k] * projW[k * 128 + c];
  out[t] = s;
}

extern "C" void kernel_launch(void* const* d_in, const int* in_sizes, int n_in,
                              void* d_out, int out_size, void* d_ws, size_t ws_size,
                              hipStream_t stream) {
  const float*      x     = (const float*)d_in[0];
  const long long*  ei    = (const long long*)d_in[1];
  const float*      ea    = (const float*)d_in[2];
  const long long*  batch = (const long long*)d_in[3];
  const float* W1l = (const float*)d_in[4];  const float* b1l = (const float*)d_in[5];
  const float* W1r = (const float*)d_in[6];  const float* b1r = (const float*)d_in[7];
  const float* W1e = (const float*)d_in[8];  const float* att1 = (const float*)d_in[9];
  const float* bias1 = (const float*)d_in[10];
  const float* ln1w = (const float*)d_in[11]; const float* ln1b = (const float*)d_in[12];
  const float* W2l = (const float*)d_in[13]; const float* b2l = (const float*)d_in[14];
  const float* W2r = (const float*)d_in[15]; const float* b2r = (const float*)d_in[16];
  const float* W2e = (const float*)d_in[17]; const float* att2 = (const float*)d_in[18];
  const float* bias2 = (const float*)d_in[19];
  const float* ln2w = (const float*)d_in[20]; const float* ln2b = (const float*)d_in[21];
  const float* projW = (const float*)d_in[22]; const float* projb = (const float*)d_in[23];
  float* out = (float*)d_out;

  // workspace layout (float units unless noted)
  float* ws = (float*)d_ws;
  const size_t NN = (size_t)N_NODESC * HC;         // 12.8M
  float*     xl     = ws;
  float*     xr     = xl + NN;
  float*     agg    = xr + NN;
  _Float16*  h1f    = (_Float16*)(agg + NN);       // NN halves = NN/2 floats
  float*     h2     = (float*)(h1f + NN);
  float*     logits = h2 + (size_t)N_NODESC * CHN; // ETOT*4
  unsigned*  lmax   = (unsigned*)(logits + (size_t)N_ETOT * HEADS);
  float*     denom  = (float*)(lmax + (size_t)N_NODESC * HEADS);
  float*     lsum   = denom + (size_t)N_NODESC * HEADS;
  float*     cntf   = lsum + (size_t)N_NODESC * 2;
  float*     lattr  = cntf + N_NODESC;
  float*     gsum   = lattr + (size_t)N_NODESC * 2;
  unsigned*  gmax   = (unsigned*)(gsum + NUM_G * CHN);
  float*     gcnt   = (float*)(gmax + NUM_G * CHN);
  _Float16*  Wlt    = (_Float16*)(gcnt + NUM_G);   // 256x256 halves
  _Float16*  Wrt    = Wlt + (size_t)HC * HC;

  const int T = 256;
  const int edgeWaveBlocks = (N_ETOT * 32 + T - 1) / T;     // 106250
  const int nodeWaveBlocks = (N_NODESC * 32 + T - 1) / T;   // 6250
  const int gemmWaves  = (N_NODESC / 16) * 4;               // 12500 waves
  const int gemmBlocks = (gemmWaves * 32 + T - 1) / T;      // 1563

  // ---- self loops + weight prep ----
  hipMemsetAsync(lsum, 0, (size_t)N_NODESC * 3 * sizeof(float), stream);  // lsum + cntf
  selfloop_accum<<<(N_EDGESC + T - 1) / T, T, 0, stream>>>(ei, ea, lsum, cntf);
  selfloop_finalize<<<(N_NODESC + T - 1) / T, T, 0, stream>>>(lsum, cntf, lattr);
  convert_w_f16<<<(HC * HC + T - 1) / T, T, 0, stream>>>(W2l, W2r, Wlt, Wrt);

  // ---- conv1 ----
  conv1_transform<<<(N_NODESC * HC + T - 1) / T, T, 0, stream>>>(x, W1l, b1l, W1r, b1r, xl, xr);
  hipMemsetAsync(lmax, 0, (size_t)N_NODESC * HEADS * 2 * sizeof(float), stream);  // lmax + denom
  hipMemsetAsync(agg, 0, NN * sizeof(float), stream);
  edge_logits<<<edgeWaveBlocks, T, 0, stream>>>(xl, xr, W1e, att1, ei, ea, lattr, logits, lmax);
  edge_exp<<<(N_ETOT * HEADS + T - 1) / T, T, 0, stream>>>(ei, lmax, logits, denom);
  edge_scatter<<<edgeWaveBlocks, T, 0, stream>>>(xl, ei, logits, denom, agg);
  node_post1<<<nodeWaveBlocks, T, 0, stream>>>(agg, bias1, ln1w, ln1b, h1f);

  // ---- conv2 ----
  gemm_dual_256<<<gemmBlocks, T, 0, stream>>>(h1f, Wlt, b2l, Wrt, b2r, xl, xr);
  hipMemsetAsync(lmax, 0, (size_t)N_NODESC * HEADS * 2 * sizeof(float), stream);
  hipMemsetAsync(agg, 0, NN * sizeof(float), stream);
  edge_logits<<<edgeWaveBlocks, T, 0, stream>>>(xl, xr, W2e, att2, ei, ea, lattr, logits, lmax);
  edge_exp<<<(N_ETOT * HEADS + T - 1) / T, T, 0, stream>>>(ei, lmax, logits, denom);
  edge_scatter<<<edgeWaveBlocks, T, 0, stream>>>(xl, ei, logits, denom, agg);
  node_post2<<<nodeWaveBlocks, T, 0, stream>>>(agg, bias2, ln2w, ln2b, h2);

  // ---- pooling + projection ----
  hipMemsetAsync(gsum, 0, (size_t)(NUM_G * CHN * 2 + NUM_G) * sizeof(float), stream);
  pool_nodes<<<nodeWaveBlocks, T, 0, stream>>>(h2, batch, gsum, gmax, gcnt);
  final_proj<<<1, 1024, 0, stream>>>(gsum, gmax, gcnt, projW, projb, out);
}
